// BidirLSTMLayer_88587995447889
// MI455X (gfx1250) — matrix operand order repaired and checked by reading, not compiled, not run on hardware
//
#include <hip/hip_runtime.h>

#define TT 1024
#define BB 64
#define II 512
#define HH 512
#define NG 2048   // 4*H gate rows
#define KK 1024   // I + H

typedef __attribute__((ext_vector_type(16))) __bf16 v16bf;
typedef __attribute__((ext_vector_type(8)))  __bf16 v8bf;
typedef __attribute__((ext_vector_type(8)))  float  v8f;
typedef unsigned short u16;
typedef unsigned int   u32;

__device__ __forceinline__ u16 f2bf(float f) {
  u32 u = __float_as_uint(f);
  u32 r = u + 0x7fffu + ((u >> 16) & 1u);   // round-to-nearest-even
  return (u16)(r >> 16);
}

// ---------------- prep kernels ----------------

// Wcat[dir][n][k] bf16, k<512 -> w_ih[n][k], else w_hh[n][k-512]
__global__ void pack_weights(const float* __restrict__ wihf, const float* __restrict__ whhf,
                             const float* __restrict__ wihb, const float* __restrict__ whhb,
                             u16* __restrict__ Wcat) {
  size_t tid = (size_t)blockIdx.x * blockDim.x + threadIdx.x;
  size_t total = (size_t)2 * NG * KK;
  if (tid >= total) return;
  int k = (int)(tid % KK);
  int n = (int)((tid / KK) % NG);
  int d = (int)(tid / ((size_t)NG * KK));
  const float* wih = d ? wihb : wihf;
  const float* whh = d ? whhb : whhf;
  float v = (k < II) ? wih[(size_t)n * II + k] : whh[(size_t)n * HH + (k - II)];
  Wcat[tid] = f2bf(v);
}

__global__ void pack_bias(const float* __restrict__ bihf, const float* __restrict__ bhhf,
                          const float* __restrict__ bihb, const float* __restrict__ bhhb,
                          float* __restrict__ bias) {
  int tid = blockIdx.x * blockDim.x + threadIdx.x;
  if (tid >= 2 * NG) return;
  int n = tid % NG;
  bias[tid] = (tid < NG) ? (bihf[n] + bhhf[n]) : (bihb[n] + bhhb[n]);
}

// Xf = bf16(x); Xr = bf16(flip_batch(x, lengths))
__global__ void conv_x(const float* __restrict__ x, const int* __restrict__ len,
                       u16* __restrict__ Xf, u16* __restrict__ Xr) {
  size_t tid = (size_t)blockIdx.x * blockDim.x + threadIdx.x;
  size_t total = (size_t)TT * BB * II;
  if (tid >= total) return;
  int k = (int)(tid % II);
  int b = (int)((tid / II) % BB);
  int t = (int)(tid / ((size_t)II * BB));
  Xf[tid] = f2bf(x[tid]);
  int L = len[b];
  int st = (t < L) ? (L - 1 - t) : t;
  Xr[tid] = f2bf(x[((size_t)st * BB + b) * II + k]);
}

// hbuf parity-0 <- bf16(h0); zero the per-direction barrier counters
__global__ void init_state(const float* __restrict__ h0f, const float* __restrict__ h0b,
                           u16* __restrict__ hbuf, u32* __restrict__ cnt) {
  int tid = blockIdx.x * blockDim.x + threadIdx.x;
  if (tid < 2) cnt[tid] = 0;
  if (tid >= 2 * BB * HH) return;
  int j = tid % HH;
  int b = (tid / HH) % BB;
  int d = tid / (BB * HH);
  const float* h0 = d ? h0b : h0f;
  hbuf[((size_t)d * 2 + 0) * (BB * HH) + b * HH + j] = f2bf(h0[b * HH + j]);
}

// ---------------- persistent bidirectional LSTM ----------------

// A operand (16x32 bf16): lane holds row (lane&15); K chunks {koff..+7} and {koff+8..+15}
__device__ __forceinline__ v16bf load_a16(const u16* row, int koff) {
  v8bf lo = *(const v8bf*)(row + koff);
  v8bf hi = *(const v8bf*)(row + koff + 8);
  return __builtin_shufflevector(lo, hi, 0,1,2,3,4,5,6,7,8,9,10,11,12,13,14,15);
}

// B operand (32x16 bf16): lane holds W-row (lane&15); 16 contiguous K values
__device__ __forceinline__ v16bf load_b16(const u16* p16) {
  const v8bf* p = (const v8bf*)p16;
  return __builtin_shufflevector(p[0], p[1], 0,1,2,3,4,5,6,7,8,9,10,11,12,13,14,15);
}

__device__ __forceinline__ float sigm(float x) { return 1.0f / (1.0f + __expf(-x)); }

// LDS layout (gate-interleaved so each lane's B working set spans only 8 KB and
// every ds_load offset is a compile-time immediate after full unroll):
//   sW[r][g][k]  : r = sub-row 0..15, g = gate 0..3, k = 0..1023 (bf16)
__global__ void __launch_bounds__(128)
lstm_persistent(const u16* __restrict__ Wcat, const float* __restrict__ bias,
                const u16* __restrict__ Xf, const u16* __restrict__ Xr,
                const float* __restrict__ c0f, const float* __restrict__ c0b,
                const int* __restrict__ len,
                u16* __restrict__ hbuf, u32* __restrict__ cnt,
                float* __restrict__ out) {
  extern __shared__ u16 sW[];            // 16 * 4 * 1024 bf16 = 128 KB
  const int dir = blockIdx.x >> 5;       // 0 = forward, 1 = backward
  const int jt  = blockIdx.x & 31;       // j-block [jt*16, jt*16+16)
  const int wv  = threadIdx.x >> 5;      // m_tile (batch tile 0..3)
  const int ln  = threadIdx.x & 31;
  const int r   = ln & 15;
  const int hf  = ln >> 4;

  // ---- stage this block's 64 gate rows of W into LDS (one time) ----
  // LDS uint4 u -> r = u>>9, g = (u>>7)&3, cu = u&127 ; global row n = g*512 + jt*16 + r
  const u16* Wd = Wcat + (size_t)dir * NG * KK;
  for (int u = threadIdx.x; u < 8192; u += 128) {   // 8192 uint4 = 128 KB
    int rr = u >> 9;
    int g  = (u >> 7) & 3;
    int cu = u & 127;
    int n  = g * HH + jt * 16 + rr;
    ((uint4*)sW)[u] = *((const uint4*)(Wd + (size_t)n * KK) + cu);
  }
  __syncthreads();

  const int j = jt * 16 + r;
  const float bI = bias[dir * NG + 0 * HH + j];
  const float bF = bias[dir * NG + 1 * HH + j];
  const float bG = bias[dir * NG + 2 * HH + j];
  const float bO = bias[dir * NG + 3 * HH + j];

  const int brow = wv * 16 + r;   // batch row feeding the A operand
  const int aoff = hf * 16;       // A half-wave K offset (folded into row ptrs)

  // single per-lane B base: r sub-row + half-wave K offset; all further offsets
  // are g*1024 + kk (compile-time after full unroll, max 8176 bytes)
  const u16* sB = sW + (r * 4096 + hf * 16);

  // persistent cell state (C-layout: vgpr v -> batch wv*16 + v + 8*hf, col j)
  float c[8];
  int Lb[8];
  const float* c0 = dir ? c0b : c0f;
  #pragma unroll
  for (int v = 0; v < 8; ++v) {
    int b = wv * 16 + v + hf * 8;
    c[v]  = c0[b * HH + j];
    Lb[v] = len[b];
  }

  const u16* X = dir ? Xr : Xf;
  u32* mycnt = cnt + dir;

  for (int t = 0; t < TT; ++t) {
    const int par = t & 1;
    const u16* hread = hbuf + (size_t)(dir * 2 + par)       * (BB * HH);
    u16*       hwr   = hbuf + (size_t)(dir * 2 + (par ^ 1)) * (BB * HH);
    const u16* xrow  = X + ((size_t)t * BB + brow) * II + aoff;
    const u16* hrow  = hread + brow * HH + aoff;

    v8f ai = {0,0,0,0,0,0,0,0}, af = {0,0,0,0,0,0,0,0};
    v8f ag = {0,0,0,0,0,0,0,0}, ao = {0,0,0,0,0,0,0,0};

    // K = [0,512): x_t contribution  (fully unrolled: immediate offsets only)
    #pragma unroll
    for (int kb = 0; kb < 16; ++kb) {
      int kk = kb * 32;
      v16bf a  = load_a16(xrow, kk);
      v16bf b0 = load_b16(sB + 0 * KK + kk);
      v16bf b1 = load_b16(sB + 1 * KK + kk);
      v16bf b2 = load_b16(sB + 2 * KK + kk);
      v16bf b3 = load_b16(sB + 3 * KK + kk);
      ai = __builtin_amdgcn_wmma_f32_16x16x32_bf16(false, a, false, b0, (short)0, ai, false, false);
      af = __builtin_amdgcn_wmma_f32_16x16x32_bf16(false, a, false, b1, (short)0, af, false, false);
      ag = __builtin_amdgcn_wmma_f32_16x16x32_bf16(false, a, false, b2, (short)0, ag, false, false);
      ao = __builtin_amdgcn_wmma_f32_16x16x32_bf16(false, a, false, b3, (short)0, ao, false, false);
    }
    // K = [512,1024): h_{t-1} contribution
    #pragma unroll
    for (int kb = 0; kb < 16; ++kb) {
      int kk = kb * 32;
      int kw = 512 + kk;
      v16bf a  = load_a16(hrow, kk);
      v16bf b0 = load_b16(sB + 0 * KK + kw);
      v16bf b1 = load_b16(sB + 1 * KK + kw);
      v16bf b2 = load_b16(sB + 2 * KK + kw);
      v16bf b3 = load_b16(sB + 3 * KK + kw);
      ai = __builtin_amdgcn_wmma_f32_16x16x32_bf16(false, a, false, b0, (short)0, ai, false, false);
      af = __builtin_amdgcn_wmma_f32_16x16x32_bf16(false, a, false, b1, (short)0, af, false, false);
      ag = __builtin_amdgcn_wmma_f32_16x16x32_bf16(false, a, false, b2, (short)0, ag, false, false);
      ao = __builtin_amdgcn_wmma_f32_16x16x32_bf16(false, a, false, b3, (short)0, ao, false, false);
    }

    // cell update + output (all four gates are element-aligned in C layout)
    #pragma unroll
    for (int v = 0; v < 8; ++v) {
      float gi = sigm(ai[v] + bI);
      float gf = sigm(af[v] + bF);
      float gg = tanhf(ag[v] + bG);
      float go = sigm(ao[v] + bO);
      float cn = gf * c[v] + gi * gg;
      c[v] = cn;
      float h = go * tanhf(cn);
      int b = wv * 16 + v + hf * 8;
      hwr[b * HH + j] = f2bf(h);
      int to = dir ? ((t < Lb[v]) ? (Lb[v] - 1 - t) : t) : t;
      out[((size_t)to * BB + b) * (2 * HH) + dir * HH + j] = h;
    }

    // per-direction grid barrier (monotonic counter, release/acquire)
    if (t != TT - 1) {
      __threadfence();
      __syncthreads();
      if (threadIdx.x == 0)
        __hip_atomic_fetch_add(mycnt, 1u, __ATOMIC_RELEASE, __HIP_MEMORY_SCOPE_AGENT);
      u32 target = 32u * (u32)(t + 1);
      while (__hip_atomic_load(mycnt, __ATOMIC_ACQUIRE, __HIP_MEMORY_SCOPE_AGENT) < target)
        __builtin_amdgcn_s_sleep(2);
      __syncthreads();
    }
  }
}

// ---------------- host launch ----------------

extern "C" void kernel_launch(void* const* d_in, const int* in_sizes, int n_in,
                              void* d_out, int out_size, void* d_ws, size_t ws_size,
                              hipStream_t stream) {
  const float* x    = (const float*)d_in[0];
  const int*   len  = (const int*)  d_in[1];
  const float* h0f  = (const float*)d_in[2];
  const float* c0f  = (const float*)d_in[3];
  const float* h0b  = (const float*)d_in[4];
  const float* c0b  = (const float*)d_in[5];
  const float* wihf = (const float*)d_in[6];
  const float* whhf = (const float*)d_in[7];
  const float* bihf = (const float*)d_in[8];
  const float* bhhf = (const float*)d_in[9];
  const float* wihb = (const float*)d_in[10];
  const float* whhb = (const float*)d_in[11];
  const float* bihb = (const float*)d_in[12];
  const float* bhhb = (const float*)d_in[13];
  float* out = (float*)d_out;

  // workspace layout (bytes), ~143 MB total
  char* w = (char*)d_ws;
  u16*   Wcat = (u16*)  (w + 0);           //  8,388,608 : 2 x 2048 x 1024 bf16
  float* bias = (float*)(w + 8388608);     //     16,384 : 2 x 2048 f32
  u16*   Xf   = (u16*)  (w + 8404992);     // 67,108,864 : T*B*I bf16
  u16*   Xr   = (u16*)  (w + 75513856);    // 67,108,864 : flipped x bf16
  u16*   hbuf = (u16*)  (w + 142622720);   //    262,144 : 2 dir x 2 parity x 64x512 bf16
  u32*   cnt  = (u32*)  (w + 142884864);   //          8 : barrier counters

  pack_weights<<<(2 * NG * KK + 255) / 256, 256, 0, stream>>>(wihf, whhf, wihb, whhb, Wcat);
  pack_bias<<<(2 * NG + 255) / 256, 256, 0, stream>>>(bihf, bhhf, bihb, bhhb, bias);
  conv_x<<<(TT * BB * II + 255) / 256, 256, 0, stream>>>(x, len, Xf, Xr);
  init_state<<<(2 * BB * HH + 255) / 256, 256, 0, stream>>>(h0f, h0b, hbuf, cnt);

  // 64 persistent blocks: dir(2) x jt(32); 128 threads = 4 waves; 128 KB dynamic LDS
  lstm_persistent<<<64, 128, 64 * KK * (int)sizeof(u16), stream>>>(
      Wcat, bias, Xf, Xr, c0f, c0b, len, hbuf, cnt, out);
}
